// DynamicGWConv1d_35244501631225
// MI455X (gfx1250) — matrix operand order repaired and checked by reading, not compiled, not run on hardware
//
#include <hip/hip_runtime.h>

// Problem constants
#define BB   16
#define CIN  64
#define COUT 64
#define HH   64
#define TT   16384
#define KK   5
#define CC   16
#define H2   128   // 2*H
#define LOG2T 14

typedef __attribute__((ext_vector_type(16))) _Float16 v16h;
typedef __attribute__((ext_vector_type(8)))  _Float16 v8h;
typedef __attribute__((ext_vector_type(8)))  float    v8f;

// Concatenate two 8-element f16 vectors into the 16-element WMMA operand.
__device__ __forceinline__ v16h cat8(v8h lo, v8h hi) {
    return __builtin_shufflevector(lo, hi, 0, 1, 2, 3, 4, 5, 6, 7,
                                           8, 9, 10, 11, 12, 13, 14, 15);
}

__device__ __forceinline__ v8f wmma16(const v16h& a, const v16h& b, const v8f& c) {
    return __builtin_amdgcn_wmma_f32_16x16x32_f16(false, a, false, b, (short)0, c, false, false);
}

// Fragment load helpers (CDNA5 ISA 7.12.2 layouts, wave32):
//  A (16x32 f16, row-major LDS [m][rowstride]): lane elements 0..7 = K base..base+7,
//  8..15 = K base+16..base+23, base = kc + 8*(lane>=16). Two contiguous 16B chunks.
__device__ __forceinline__ v16h load_a_frag(const _Float16* row, int kc, int lane) {
    const v8h* p = (const v8h*)(row + kc + ((lane & 16) ? 8 : 0));
    return cat8(p[0], p[2]);
}
//  B (32x16 f16) with LDS stored transposed [n][k]: lane elements 0..7 = K base..base+7,
//  8..15 = K base+16..base+23, base = kc + 8*(lane>=16). Two 16B chunks.
__device__ __forceinline__ v16h load_b_frag(const _Float16* row, int kc, int lane) {
    const v8h* p = (const v8h*)(row + kc + ((lane & 16) ? 8 : 0));
    return cat8(p[0], p[2]);
}

// ============================================================================
// Kernel 1: res = conv_w @ x + conv_b   (-> d_out)
//           e   = emb_w  @ x + emb_b    (-> e_buf)
// 8 waves: sel = w/4 (0=res,1=emb), mtile = w%4; each wave 16x64 out, K=64.
// ============================================================================
__global__ __launch_bounds__(256) void k_in_gemm(
    const float* __restrict__ x, const float* __restrict__ conv_w,
    const float* __restrict__ conv_b, const float* __restrict__ emb_w,
    const float* __restrict__ emb_b, float* __restrict__ out_res,
    float* __restrict__ e_buf)
{
    __shared__ __align__(16) _Float16 Ws16[2 * 64 * 64];   // [sel][m][k]
    __shared__ __align__(16) _Float16 Xs16[64 * 64];       // transposed [n][k]
    const int b = blockIdx.y, col0 = blockIdx.x * 64, tid = threadIdx.x;

    for (int i = tid; i < 64 * 64; i += 256) {
        Ws16[i]        = (_Float16)conv_w[i];
        Ws16[4096 + i] = (_Float16)emb_w[i];
        int c = i >> 6, n = i & 63;
        Xs16[n * 64 + c] = (_Float16)x[(long)((b << 6) + c) * TT + col0 + n];
    }
    __syncthreads();

    const int lane = tid & 31, w = tid >> 5;
    const int sel = w >> 2, mtile = w & 3;
    const _Float16* Wsel = Ws16 + sel * 4096;

    v8f acc[4] = {};
    for (int kc = 0; kc < 64; kc += 32) {
        const int m = mtile * 16 + (lane & 15);
        v16h af = load_a_frag(Wsel + m * 64, kc, lane);
#pragma unroll
        for (int nt = 0; nt < 4; ++nt) {
            const int n = nt * 16 + (lane & 15);
            v16h bf = load_b_frag(Xs16 + n * 64, kc, lane);
            acc[nt] = wmma16(af, bf, acc[nt]);
        }
    }

    const float* bias = sel ? emb_b : conv_b;
    float* dst = sel ? e_buf : out_res;
#pragma unroll
    for (int nt = 0; nt < 4; ++nt)
#pragma unroll
        for (int r = 0; r < 8; ++r) {
            int m = mtile * 16 + r + ((lane >= 16) ? 8 : 0);
            int n = col0 + nt * 16 + (lane & 15);
            dst[(long)((b << 6) + m) * TT + n] = acc[nt][r] + bias[m];
        }
}

// ============================================================================
// Kernel 2: forward FFT (one 16384-pt row per block, radix-2 DIT in 128KB LDS),
// save DC real, multiply by 1j*freq, fftshift, write Xr/Xi.
// ============================================================================
__global__ __launch_bounds__(256) void k_fft_fwd(
    const float* __restrict__ e_in, float* __restrict__ Xr,
    float* __restrict__ Xi, float* __restrict__ dc)
{
    extern __shared__ float sm[];
    float* re = sm;
    float* im = sm + TT;
    const int row = blockIdx.x, tid = threadIdx.x;
    const float* src = e_in + (long)row * TT;

    for (int i = tid; i < TT; i += 256) {
        int j = (int)(__brev((unsigned)i) >> (32 - LOG2T));
        re[j] = src[i];
        im[j] = 0.f;
    }
    __syncthreads();

    for (int sh = 0; sh < LOG2T; ++sh) {
        const int half = 1 << sh, len = 2 << sh;
        const float base = -6.28318530717958647692f / (float)len;
        for (int idx = tid; idx < (TT >> 1); idx += 256) {
            int blk = idx >> sh, pos = idx & (half - 1);
            int i0 = blk * len + pos, i1 = i0 + half;
            float s, c;
            __sincosf(base * (float)pos, &s, &c);
            float r1 = re[i1], m1 = im[i1];
            float tr = c * r1 - s * m1, ti = c * m1 + s * r1;
            float r0 = re[i0], m0 = im[i0];
            re[i1] = r0 - tr; im[i1] = m0 - ti;
            re[i0] = r0 + tr; im[i0] = m0 + ti;
        }
        __syncthreads();
    }

    if (tid == 0) dc[row] = re[0];

    for (int kf = tid; kf < TT; kf += 256) {
        float fr = ((kf < (TT / 2)) ? (float)kf : (float)(kf - TT)) * (1.0f / (float)TT);
        float xr = re[kf], xi = im[kf];
        long o = (long)row * TT + (kf ^ (TT / 2));   // fftshift
        Xr[o] = -fr * xi;                            // X *= 1j*freq
        Xi[o] =  fr * xr;
    }
}

// ============================================================================
// Kernel 3: attention head (tiny). One block (64 thr) per batch.
// ============================================================================
__global__ __launch_bounds__(64) void k_attn(
    const float* __restrict__ dc, const float* __restrict__ w1,
    const float* __restrict__ b1, const float* __restrict__ w2,
    const float* __restrict__ b2, float* __restrict__ attn)
{
    __shared__ float a_s[HH];
    __shared__ float l_s[CC];
    const int b = blockIdx.x, tid = threadIdx.x;

    float acc = b1[tid];
    for (int c = 0; c < HH; ++c) acc += dc[b * HH + c] * w1[tid * HH + c];
    a_s[tid] = fmaxf(acc, 0.f);
    __syncthreads();

    if (tid < CC) {
        float l = b2[tid];
        for (int j = 0; j < HH; ++j) l += a_s[j] * w2[tid * HH + j];
        l_s[tid] = l;
    }
    __syncthreads();

    if (tid == 0) {
        float mx = l_s[0];
        for (int c = 1; c < CC; ++c) mx = fmaxf(mx, l_s[c]);
        float sum = 0.f, ex[CC];
        for (int c = 0; c < CC; ++c) { ex[c] = __expf(l_s[c] - mx); sum += ex[c]; }
        float inv = 1.0f / sum;
        for (int c = 0; c < CC; ++c) attn[b * CC + c] = ex[c] * inv;
    }
}

// ============================================================================
// Kernel 4: kmix[b,k,o,h] = sum_c ker[o,h,k,c] * attn[b,c]   (complex)
// ============================================================================
__global__ __launch_bounds__(256) void k_kmix(
    const float* __restrict__ kr, const float* __restrict__ ki,
    const float* __restrict__ attn, float* __restrict__ kmr,
    float* __restrict__ kmi)
{
    const int idx = blockIdx.x * 256 + threadIdx.x;   // total B*K*128*64 = 655360
    const int h = idx & 63;
    int rem = idx >> 6;
    const int o = rem & 127;
    rem >>= 7;
    const int k = rem % KK, b = rem / KK;
    const float* aw = attn + b * CC;
    const int kbase = ((o * HH + h) * KK + k) * CC;   // (2H,H,K,1,1,C) strides
    float sr = 0.f, si = 0.f;
    for (int c = 0; c < CC; ++c) {
        float w = aw[c];
        sr += kr[kbase + c] * w;
        si += ki[kbase + c] * w;
    }
    kmr[idx] = sr;
    kmi[idx] = si;
}

// ============================================================================
// Kernel 5 (dominant): per (b, tap k):
//   Y[b](128xT, cplx) += Kmix[b,k](128x64, cplx) @ Xshift_k[b](64xT, cplx)
// Block: batch b x 64 time-cols x all 128 rows. 8 waves, wave = M-tile, 4
// N-tiles each; accumulators persist across the 5 taps.
// f16 LDS, fragment-native layouts: 2x ds_load_b128 per WMMA operand, no
// per-use cvt. Shifted-spectrum tile staged ONCE with a 68-wide halo
// (transposed [j][h], tap shift = row offset). A tile restaged per tap with
// -Ai precomputed (NEG[0] must be 0 for f16 WMMA). Next tap's A prefetched
// (global_prefetch_b8).
// ============================================================================
__global__ __launch_bounds__(256) void k_freq_conv(
    const float* __restrict__ Xr, const float* __restrict__ Xi,
    const float* __restrict__ kmr, const float* __restrict__ kmi,
    float* __restrict__ Yr, float* __restrict__ Yi)
{
    extern __shared__ _Float16 smh[];
    _Float16* Ar = smh;              // 128*64  [m][k]
    _Float16* Ai = smh + 8192;       // 128*64
    _Float16* An = smh + 16384;      // 128*64  (-Ai)
    _Float16* Br = smh + 24576;      // 68*64   [j][h], j = halo column
    _Float16* Bi = Br + 68 * 64;     // 68*64
    const int b = blockIdx.y, col0 = blockIdx.x * 64, tid = threadIdx.x;
    const int lane = tid & 31, w = tid >> 5;   // mtile = w

    // Stage halo spectrum tile once: j in [0,68) covers global col (col0+j-2).
    for (int i = tid; i < 68 * 64; i += 256) {
        int h = i / 68, j = i - h * 68;
        int g = col0 + j - (KK / 2);
        float vr = 0.f, vi = 0.f;
        if ((unsigned)g < (unsigned)TT) {
            long o = ((long)(b * HH + h) << LOG2T) + g;
            vr = Xr[o]; vi = Xi[o];
        }
        Br[j * 64 + h] = (_Float16)vr;
        Bi[j * 64 + h] = (_Float16)vi;
    }

    v8f accr[4] = {};
    v8f acci[4] = {};

    for (int k = 0; k < KK; ++k) {
        __syncthreads();   // also covers initial B staging
        // stage A = kmix[b][k] (128x64 complex) as f16, with negated imag
        const float* arg = kmr + (long)(b * KK + k) * 8192;
        const float* aig = kmi + (long)(b * KK + k) * 8192;
        for (int i = tid; i < 8192; i += 256) {
            Ar[i] = (_Float16)arg[i];
            _Float16 hv = (_Float16)aig[i];
            Ai[i] = hv;
            An[i] = -hv;
        }
        if (k + 1 < KK) {   // prefetch next tap's A into cache
            const float* nrg = kmr + (long)(b * KK + k + 1) * 8192;
            const float* nig = kmi + (long)(b * KK + k + 1) * 8192;
            __builtin_prefetch(nrg + tid * 32, 0, 1);
            __builtin_prefetch(nig + tid * 32, 0, 1);
        }
        __syncthreads();

        for (int kc = 0; kc < 64; kc += 32) {
            const int m = (w << 4) + (lane & 15);
            v16h far = load_a_frag(Ar + m * 64, kc, lane);
            v16h fai = load_a_frag(Ai + m * 64, kc, lane);
            v16h fan = load_a_frag(An + m * 64, kc, lane);
#pragma unroll
            for (int nt = 0; nt < 4; ++nt) {
                const int j = (nt << 4) + (lane & 15) + k;   // halo row
                v16h fbr = load_b_frag(Br + j * 64, kc, lane);
                v16h fbi = load_b_frag(Bi + j * 64, kc, lane);
                accr[nt] = wmma16(far, fbr, accr[nt]);   // + Ar*Xr
                accr[nt] = wmma16(fan, fbi, accr[nt]);   // - Ai*Xi
                acci[nt] = wmma16(far, fbi, acci[nt]);   // + Ar*Xi
                acci[nt] = wmma16(fai, fbr, acci[nt]);   // + Ai*Xr
            }
        }
    }

#pragma unroll
    for (int nt = 0; nt < 4; ++nt)
#pragma unroll
        for (int r = 0; r < 8; ++r) {
            int o = (w << 4) + r + ((lane >= 16) ? 8 : 0);
            int n = col0 + (nt << 4) + (lane & 15);
            long idx = ((long)(b * H2 + o) << LOG2T) + n;
            Yr[idx] = accr[nt][r];
            Yi[idx] = acci[nt][r];
        }
}

// ============================================================================
// Kernel 6: inverse FFT per row (ifftshift folded into load), real part * 1/T.
// ============================================================================
__global__ __launch_bounds__(256) void k_fft_inv(
    const float* __restrict__ Yr, const float* __restrict__ Yi,
    float* __restrict__ yout)
{
    extern __shared__ float sm[];
    float* re = sm;
    float* im = sm + TT;
    const int row = blockIdx.x, tid = threadIdx.x;

    for (int i = tid; i < TT; i += 256) {
        int j = (int)(__brev((unsigned)i) >> (32 - LOG2T));
        long o = (long)row * TT + (i ^ (TT / 2));   // ifftshift
        re[j] = Yr[o];
        im[j] = Yi[o];
    }
    __syncthreads();

    for (int sh = 0; sh < LOG2T; ++sh) {
        const int half = 1 << sh, len = 2 << sh;
        const float base = 6.28318530717958647692f / (float)len;  // conj twiddle
        for (int idx = tid; idx < (TT >> 1); idx += 256) {
            int blk = idx >> sh, pos = idx & (half - 1);
            int i0 = blk * len + pos, i1 = i0 + half;
            float s, c;
            __sincosf(base * (float)pos, &s, &c);
            float r1 = re[i1], m1 = im[i1];
            float tr = c * r1 - s * m1, ti = c * m1 + s * r1;
            float r0 = re[i0], m0 = im[i0];
            re[i1] = r0 - tr; im[i1] = m0 - ti;
            re[i0] = r0 + tr; im[i0] = m0 + ti;
        }
        __syncthreads();
    }

    const float scale = 1.0f / (float)TT;
    for (int t = tid; t < TT; t += 256)
        yout[(long)row * TT + t] = re[t] * scale;
}

// ============================================================================
// Kernel 7: d_out += 0.001 * (proj_w(64x128) @ y[b](128xT) + proj_b)
// 8 waves: mtile = w%4, N-tile pair w/4. K=128 via 4 WMMA chunks. f16 LDS.
// ============================================================================
__global__ __launch_bounds__(256) void k_proj(
    const float* __restrict__ y, const float* __restrict__ pw,
    const float* __restrict__ pb, float* __restrict__ out)
{
    __shared__ __align__(16) _Float16 Ws16[64 * 128];   // [m][c]
    __shared__ __align__(16) _Float16 Ys16[64 * 128];   // transposed [n][c]
    const int b = blockIdx.y, col0 = blockIdx.x * 64, tid = threadIdx.x;

    for (int i = tid; i < 8192; i += 256) {
        Ws16[i] = (_Float16)pw[i];
        int c = i >> 6, n = i & 63;
        Ys16[n * 128 + c] = (_Float16)y[((long)(b * H2 + c) << LOG2T) + col0 + n];
    }
    __syncthreads();

    const int lane = tid & 31, w = tid >> 5;
    const int mtile = w & 3, npair = w >> 2;

    v8f acc[2] = {};
    for (int kc = 0; kc < H2; kc += 32) {
        const int m = mtile * 16 + (lane & 15);
        v16h af = load_a_frag(Ws16 + m * 128, kc, lane);
#pragma unroll
        for (int j = 0; j < 2; ++j) {
            const int n = (npair * 2 + j) * 16 + (lane & 15);
            v16h bf = load_b_frag(Ys16 + n * 128, kc, lane);
            acc[j] = wmma16(af, bf, acc[j]);
        }
    }

#pragma unroll
    for (int j = 0; j < 2; ++j) {
        int nt = npair * 2 + j;
#pragma unroll
        for (int r = 0; r < 8; ++r) {
            int m = mtile * 16 + r + ((lane >= 16) ? 8 : 0);
            int n = col0 + nt * 16 + (lane & 15);
            long o = ((long)(b * COUT + m) << LOG2T) + n;
            out[o] += 0.001f * (acc[j][r] + pb[m]);
        }
    }
}

// ============================================================================
extern "C" void kernel_launch(void* const* d_in, const int* in_sizes, int n_in,
                              void* d_out, int out_size, void* d_ws, size_t ws_size,
                              hipStream_t stream)
{
    (void)in_sizes; (void)n_in; (void)out_size; (void)ws_size;
    const float* x       = (const float*)d_in[0];
    const float* conv_w  = (const float*)d_in[1];
    const float* conv_b  = (const float*)d_in[2];
    const float* emb_w   = (const float*)d_in[3];
    const float* emb_b   = (const float*)d_in[4];
    const float* attn_w1 = (const float*)d_in[5];
    const float* attn_b1 = (const float*)d_in[6];
    const float* attn_w2 = (const float*)d_in[7];
    const float* attn_b2 = (const float*)d_in[8];
    const float* ker_r   = (const float*)d_in[9];
    const float* ker_i   = (const float*)d_in[10];
    const float* proj_w  = (const float*)d_in[11];
    const float* proj_b  = (const float*)d_in[12];
    float* out = (float*)d_out;
    float* ws  = (float*)d_ws;

    const long NE = (long)BB * HH * TT;        // 16.78M floats
    float* e_buf = ws;                         // NE
    float* Xr    = ws + NE;                    // NE
    float* Xi    = ws + 2 * NE;                // NE
    float* Yr    = ws + 3 * NE;                // 2*NE
    float* Yi    = ws + 5 * NE;                // 2*NE
    float* yre   = ws + NE;                    // aliases Xr/Xi (consumed before)
    float* dc    = ws + 7 * NE;                // B*H
    float* attn  = dc + BB * HH;               // B*C
    float* kmr   = attn + BB * CC;             // B*K*128*64
    float* kmi   = kmr + (long)BB * KK * H2 * HH;

    const dim3 blk(256);
    k_in_gemm  <<<dim3(TT / 64, BB), blk, 0,          stream>>>(x, conv_w, conv_b, emb_w, emb_b, out, e_buf);
    k_fft_fwd  <<<dim3(BB * HH),     blk, 2 * TT * 4, stream>>>(e_buf, Xr, Xi, dc);
    k_attn     <<<dim3(BB), dim3(64), 0,              stream>>>(dc, attn_w1, attn_b1, attn_w2, attn_b2, attn);
    k_kmix     <<<dim3((BB * KK * H2 * HH) / 256), blk, 0, stream>>>(ker_r, ker_i, attn, kmr, kmi);
    k_freq_conv<<<dim3(TT / 64, BB), blk, 66560,      stream>>>(Xr, Xi, kmr, kmi, Yr, Yi);
    k_fft_inv  <<<dim3(BB * H2),     blk, 2 * TT * 4, stream>>>(Yr, Yi, yre);
    k_proj     <<<dim3(TT / 64, BB), blk, 0,          stream>>>(yre, proj_w, proj_b, out);
}
